// BinConvBnHTanh_64364379898573
// MI455X (gfx1250) — 8-bit matrix operand order repaired and checked by reading, compiled, not run on hardware
//
#include <hip/hip_runtime.h>


typedef __attribute__((ext_vector_type(8))) int v8i;

union V8 { v8i v; int4 q[2]; };

#define CIN    64
#define COUT   64
#define HW     256
#define LDSCOL 132
#define AFRAG_BYTES (4*9*32*32)

__global__ __launch_bounds__(256) void prep_wfrag_kernel(
    const float* __restrict__ w, int* __restrict__ afrag)
{
    int t = blockIdx.x * 256 + threadIdx.x;
    int v     = t & 7;
    int lane  = (t >> 3) & 31;
    int chunk = t >> 8;
    int kc = chunk % 9;
    int mt = chunk / 9;
    int M  = lane & 15;
    int hi = lane >> 4;
    int Kst = ((v >> 2) << 5) + ((v & 3) << 2) + (hi << 4);
    int kh = kc / 3, kw = kc - kh * 3;
    int co = mt * 16 + M;
    unsigned d = 0;
#pragma unroll
    for (int j = 0; j < 4; ++j) {
        int ci = Kst + j;
        float wv = w[((co * CIN + ci) * 3 + kh) * 3 + kw];
        unsigned byte = (wv >= 0.0f) ? 0x01u : 0xFFu;
        d |= byte << (8 * j);
    }
    afrag[t] = (int)d;
}

__global__ __launch_bounds__(256) void binconv_wmma_kernel(
    const float* __restrict__ x,     const int* __restrict__ afrag,
    const float* __restrict__ bias,  const float* __restrict__ gamma,
    const float* __restrict__ beta,  const float* __restrict__ mean,
    const float* __restrict__ var,   float* __restrict__ out)
{
    __shared__ __align__(16) signed char xsb[6 * LDSCOL * CIN];
    __shared__ __align__(16) signed char sA[AFRAG_BYTES];
    __shared__ float s_scale[COUT], s_shift[COUT], s_bias[COUT];

    const int tid  = threadIdx.x;
    const int bx   = blockIdx.x;
    const int half = bx & 1;
    const int rq   = (bx >> 1) & 63;
    const int n    = bx >> 7;
    const int h0   = rq << 2;
    const int wb   = half * 128;

    {
        unsigned sA_base = (unsigned)(size_t)&sA[0];
#pragma unroll
        for (int i = 0; i < 9; ++i) {
            unsigned off = (unsigned)(tid + i * 256) * 16u;
            unsigned lds = sA_base + off;
            asm volatile("global_load_async_to_lds_b128 %0, %1, %2"
                         :: "v"(lds), "v"(off), "s"(afrag) : "memory");
        }
    }

    if (tid < COUT) {
        float inv = gamma[tid] * rsqrtf(var[tid] + 1e-5f);
        s_scale[tid] = inv;
        s_shift[tid] = beta[tid] - mean[tid] * inv;
        s_bias[tid]  = bias[tid];
    }

    const int TOT = 6 * 16 * 130;
    for (int idx = tid; idx < TOT; idx += 256) {
        int r   = idx / (16 * 130);
        int rem = idx - r * (16 * 130);
        int cg  = rem / 130;
        int col = rem - cg * 130;
        int hin = h0 - 1 + r;
        int win = wb - 1 + col;
        unsigned d = 0;
        if ((unsigned)hin < (unsigned)HW && (unsigned)win < (unsigned)HW) {
            const float* xp = x + ((((long)n * CIN + (cg << 2)) * HW + hin) * HW + win);
#pragma unroll
            for (int j = 0; j < 4; ++j) {
                float xv = xp[(long)j * HW * HW];
                unsigned byte = (xv >= 0.0f) ? 0x01u : 0xFFu;
                d |= byte << (8 * j);
            }
        }
        *(unsigned*)&xsb[(r * LDSCOL + col) * CIN + (cg << 2)] = d;
    }

    asm volatile("s_wait_asynccnt 0x0" ::: "memory");
    __syncthreads();

    const int lane = tid & 31;
    const int wave = tid >> 5;
    const int pxb  = wave << 4;
    const int ncol = lane & 15;
    const int hofs = (lane >> 4) << 4;
    const int wcol  = wb + pxb + ncol;
    const int coofs = (lane >> 4) << 3;

#pragma unroll 1
    for (int ry = 0; ry < 4; ++ry) {
        v8i acc0 = {0,0,0,0,0,0,0,0};
        v8i acc1 = {0,0,0,0,0,0,0,0};
        v8i acc2 = {0,0,0,0,0,0,0,0};
        v8i acc3 = {0,0,0,0,0,0,0,0};

        for (int kc = 0; kc < 9; ++kc) {
            int kh = kc / 3, kw = kc - kh * 3;

            V8 bfrag;
            const signed char* bcol =
                &xsb[((ry + kh) * LDSCOL + pxb + ncol + kw) * CIN];
            bfrag.q[0] = *(const int4*)(bcol + hofs);
            bfrag.q[1] = *(const int4*)(bcol + 32 + hofs);

            V8 a0, a1, a2, a3;
            const int4* ap = (const int4*)&sA[((kc * 32 + lane) << 5)];
            a0.q[0] = ap[0 * 576];  a0.q[1] = ap[0 * 576 + 1];
            a1.q[0] = ap[1 * 576];  a1.q[1] = ap[1 * 576 + 1];
            a2.q[0] = ap[2 * 576];  a2.q[1] = ap[2 * 576 + 1];
            a3.q[0] = ap[3 * 576];  a3.q[1] = ap[3 * 576 + 1];

            acc0 = __builtin_amdgcn_wmma_i32_16x16x64_iu8(true, a0.v, true, bfrag.v, acc0, false, false);
            acc1 = __builtin_amdgcn_wmma_i32_16x16x64_iu8(true, a1.v, true, bfrag.v, acc1, false, false);
            acc2 = __builtin_amdgcn_wmma_i32_16x16x64_iu8(true, a2.v, true, bfrag.v, acc2, false, false);
            acc3 = __builtin_amdgcn_wmma_i32_16x16x64_iu8(true, a3.v, true, bfrag.v, acc3, false, false);
        }

        const int h = h0 + ry;
#pragma unroll
        for (int r = 0; r < 8; ++r) {
            float vals[4] = { (float)acc0[r], (float)acc1[r],
                              (float)acc2[r], (float)acc3[r] };
#pragma unroll
            for (int mt = 0; mt < 4; ++mt) {
                int co = mt * 16 + coofs + r;
                float y = vals[mt] + s_bias[co];
                y = fmaxf(y, 0.0f);
                y = y * s_scale[co] + s_shift[co];
                out[(((long)n * COUT + co) * HW + h) * HW + wcol] = y;
            }
        }
    }
}

extern "C" void kernel_launch(void* const* d_in, const int* in_sizes, int n_in,
                              void* d_out, int out_size, void* d_ws, size_t ws_size,
                              hipStream_t stream) {
    (void)in_sizes; (void)n_in; (void)out_size; (void)ws_size;
    const float* x     = (const float*)d_in[0];
    const float* w     = (const float*)d_in[1];
    const float* b     = (const float*)d_in[2];
    const float* gamma = (const float*)d_in[3];
    const float* beta  = (const float*)d_in[4];
    const float* rmean = (const float*)d_in[5];
    const float* rvar  = (const float*)d_in[6];
    float* out  = (float*)d_out;
    int* afrag  = (int*)d_ws;

    prep_wfrag_kernel<<<36, 256, 0, stream>>>(w, afrag);
    binconv_wmma_kernel<<<16 * 64 * 2, 256, 0, stream>>>(
        x, afrag, b, gamma, beta, rmean, rvar, out);
}
